// GATLayer_23914377904264
// MI455X (gfx1250) — matrix operand order repaired and checked by reading, not compiled, not run on hardware
//
#include <hip/hip_runtime.h>
#include <hip/hip_bf16.h>

typedef __attribute__((ext_vector_type(16))) __bf16 v16bf;
typedef __attribute__((ext_vector_type(8)))  __bf16 v8bf;
typedef __attribute__((ext_vector_type(8)))  float  v8f;
typedef int v4i __attribute__((vector_size(16)));    // matches builtin param type

#define BATCH 8
#define SEQ   1024
#define INDIM 768
#define NHEAD 8
#define FDIM  64
#define NCOL  512      // NHEAD*FDIM
#define ALPHA 0.2f

// LDS column stride: 64B of data (32 bf16) + 16B pad -> bank-conflict-free,
// keeps 16B alignment for b128 transfers.
#define COLB 80

#if defined(__AMDGCN__) && __has_builtin(__builtin_amdgcn_global_load_async_to_lds_b128)
#define HAS_ASYNC_LDS 1
#endif

#define AS1 __attribute__((address_space(1)))
#define AS3 __attribute__((address_space(3)))

__device__ __forceinline__ void async_copy16(const void* g, void* l) {
#ifdef HAS_ASYNC_LDS
    __builtin_amdgcn_global_load_async_to_lds_b128((AS1 v4i*)g, (AS3 v4i*)l, 0, 0);
#else
    *(float4*)l = *(const float4*)g;     // synchronous fallback
#endif
}
__device__ __forceinline__ void wait_async() {
#ifdef HAS_ASYNC_LDS
    asm volatile("s_wait_asynccnt 0x0" ::: "memory");
#endif
}

// ---------------------------------------------------------------------------
// Kernel 1: W (768x512 f32, K x N row-major) -> WT (512x768 bf16, N-major)
// ---------------------------------------------------------------------------
__global__ __launch_bounds__(256) void k_wt(const float* __restrict__ W,
                                            __bf16* __restrict__ WT) {
    int tid = blockIdx.x * blockDim.x + threadIdx.x;   // over 768*512
    int k = tid / NCOL;
    int n = tid - k * NCOL;
    WT[(size_t)n * INDIM + k] = (__bf16)W[tid];
}

// ---------------------------------------------------------------------------
// Kernel 2: Wh = X @ W via bf16 WMMA, async-LDS double-buffered B operand.
// Block = 8 waves = 2 mtiles x 4 n-groups covering a 256-column window.
// Output transposed: WhT[(b*NCOL+col)*SEQ + s] (bf16).
// ---------------------------------------------------------------------------
__global__ __launch_bounds__(256) void k_gemm(const float* __restrict__ X,
                                              const __bf16* __restrict__ WT,
                                              __bf16* __restrict__ WhT) {
    __shared__ char sB[2 * 256 * COLB];              // 40 KB, double buffered

    const int tid  = threadIdx.x;
    const int wave = tid >> 5;
    const int lane = tid & 31;
    const int l  = lane & 15;
    const int hi = lane >> 4;

    const int mpair = blockIdx.x >> 1;               // 0..255
    const int nhalf = blockIdx.x & 1;                // 0..1
    const int mtile = mpair * 2 + (wave >> 2);       // 0..511
    const int nbase = nhalf * 256;                   // block's column window
    const int n0l   = (wave & 3) * 64;               // wave's cols in window

    const float* xrow = X + (size_t)(mtile * 16 + l) * INDIM;

    // Stage 256 cols x 32 K of WT into LDS buffer `buf`: 1024 x 16B chunks.
    auto stage = [&](int buf, int k0) {
#pragma unroll
        for (int i = 0; i < 4; ++i) {
            int c    = tid + i * 256;
            int col  = c >> 2;
            int part = c & 3;
            const __bf16* g = WT + (size_t)(nbase + col) * INDIM + k0 + part * 8;
            char* lp = sB + buf * (256 * COLB) + col * COLB + part * 16;
            async_copy16(g, lp);
        }
    };

    v8f acc0 = {}, acc1 = {}, acc2 = {}, acc3 = {};

    stage(0, 0);
    wait_async();
    __syncthreads();
    int buf = 0;

    for (int k0 = 0; k0 < INDIM; k0 += 32) {
        if (k0 + 32 < INDIM) stage(buf ^ 1, k0 + 32);

        // ---- A tile: 16x32 bf16, contiguous K per lane half (from global f32) ----
        float4 a0 = *(const float4*)(xrow + k0 + hi * 16);
        float4 a1 = *(const float4*)(xrow + k0 + hi * 16 + 4);
        float4 a2 = *(const float4*)(xrow + k0 + hi * 16 + 8);
        float4 a3 = *(const float4*)(xrow + k0 + hi * 16 + 12);
        v16bf A;
        A[0]=(__bf16)a0.x;  A[1]=(__bf16)a0.y;  A[2]=(__bf16)a0.z;  A[3]=(__bf16)a0.w;
        A[4]=(__bf16)a1.x;  A[5]=(__bf16)a1.y;  A[6]=(__bf16)a1.z;  A[7]=(__bf16)a1.w;
        A[8]=(__bf16)a2.x;  A[9]=(__bf16)a2.y;  A[10]=(__bf16)a2.z; A[11]=(__bf16)a2.w;
        A[12]=(__bf16)a3.x; A[13]=(__bf16)a3.y; A[14]=(__bf16)a3.z; A[15]=(__bf16)a3.w;

        // ---- B tiles from LDS (lane = N column, 16 consecutive K) ----
        const char* bb = sB + buf * (256 * COLB) + (n0l + l) * COLB + hi * 32;
        v16bf B0 = *(const v16bf*)(bb + 0 * 16 * COLB);
        v16bf B1 = *(const v16bf*)(bb + 1 * 16 * COLB);
        v16bf B2 = *(const v16bf*)(bb + 2 * 16 * COLB);
        v16bf B3 = *(const v16bf*)(bb + 3 * 16 * COLB);

        acc0 = __builtin_amdgcn_wmma_f32_16x16x32_bf16(false, A, false, B0, (short)0, acc0, false, false);
        acc1 = __builtin_amdgcn_wmma_f32_16x16x32_bf16(false, A, false, B1, (short)0, acc1, false, false);
        acc2 = __builtin_amdgcn_wmma_f32_16x16x32_bf16(false, A, false, B2, (short)0, acc2, false, false);
        acc3 = __builtin_amdgcn_wmma_f32_16x16x32_bf16(false, A, false, B3, (short)0, acc3, false, false);

        wait_async();
        __syncthreads();
        buf ^= 1;
    }

    // ---- epilogue: lane holds col n, VGPR r = row hi*8+r -> contiguous in s
    int b    = (mtile * 16) / SEQ;
    int sloc = ((mtile * 16) & (SEQ - 1)) + hi * 8;
    v8f accs[4] = {acc0, acc1, acc2, acc3};
#pragma unroll
    for (int c = 0; c < 4; ++c) {
        int col = nbase + n0l + c * 16 + l;
        v8bf o;
#pragma unroll
        for (int r = 0; r < 8; ++r) o[r] = (__bf16)accs[c][r];
        *(v8bf*)(WhT + ((size_t)(b * NCOL + col)) * SEQ + sloc) = o;
    }
}

// ---------------------------------------------------------------------------
// Kernel 3: src/dst[b,h,s] = sum_f WhT[b,h,f,s] * a[f] / a[64+f]
// ---------------------------------------------------------------------------
__global__ __launch_bounds__(256) void k_srcdst(const __bf16* __restrict__ WhT,
                                                const float* __restrict__ avec,
                                                float* __restrict__ srcb,
                                                float* __restrict__ dstb) {
    int tid = blockIdx.x * blockDim.x + threadIdx.x;   // (b*NHEAD+h)*SEQ + s
    int s  = tid & (SEQ - 1);
    int bh = tid >> 10;
    const __bf16* base = WhT + ((size_t)bh * FDIM) * SEQ + s;
    float sa = 0.f, da = 0.f;
#pragma unroll 8
    for (int f = 0; f < FDIM; ++f) {
        float w = (float)base[(size_t)f * SEQ];
        sa += w * avec[f];
        da += w * avec[FDIM + f];
    }
    srcb[tid] = sa;
    dstb[tid] = da;
}

// ---------------------------------------------------------------------------
// Kernel 4: fused flash-GAT attention with async-LDS staged Wh operand.
// All 8 waves of a block share one (b,h); the 64x32 bf16 Wh slab per j-chunk
// is staged once into LDS (double buffered) and consumed by all waves.
// ---------------------------------------------------------------------------
__global__ __launch_bounds__(256) void k_attn(const __bf16* __restrict__ WhT,
                                              const float* __restrict__ srcb,
                                              const float* __restrict__ dstb,
                                              const int* __restrict__ mask,
                                              float* __restrict__ out) {
    __shared__ char sS[2 * 64 * COLB];               // 10 KB, double buffered

    const int tid  = threadIdx.x;
    const int wave = tid >> 5;
    const int lane = tid & 31;
    const int l  = lane & 15;
    const int hi = lane >> 4;

    const int bh    = blockIdx.x >> 3;               // 0..63, same for block
    const int itile = (blockIdx.x & 7) * 8 + wave;   // 0..63
    const int b = bh >> 3;
    const int h = bh & 7;

    float si = srcb[bh * SEQ + itile * 16 + l];      // this lane's row logit
    const float*  drow  = dstb + bh * SEQ;
    const int*    mrow  = mask + b * SEQ;
    const __bf16* Bglob = WhT + ((size_t)bh * FDIM) * SEQ;

    // Stage 64 f-cols x 32 j of Wh into LDS buffer `buf`: 256 x 16B chunks.
    auto stage = [&](int buf, int j0) {
        int f    = tid >> 2;
        int part = tid & 3;
        const __bf16* g = Bglob + (size_t)f * SEQ + j0 + part * 8;
        char* lp = sS + buf * (64 * COLB) + f * COLB + part * 16;
        async_copy16(g, lp);
    };

    v8f acc0 = {}, acc1 = {}, acc2 = {}, acc3 = {};
    float m = -INFINITY, sum = 0.f;

    stage(0, 0);
    wait_async();
    __syncthreads();
    int buf = 0;

    for (int j0 = 0; j0 < SEQ; j0 += 32) {
        if (j0 + 32 < SEQ) stage(buf ^ 1, j0 + 32);

        int ja = j0 + hi * 16;       // contiguous K: elements 0-7 for this lane
        int jb = j0 + hi * 16 + 8;   // contiguous K: elements 8-15

        float4 d0 = *(const float4*)(drow + ja);
        float4 d1 = *(const float4*)(drow + ja + 4);
        float4 d2 = *(const float4*)(drow + jb);
        float4 d3 = *(const float4*)(drow + jb + 4);
        int4 k0 = *(const int4*)(mrow + ja);
        int4 k1 = *(const int4*)(mrow + ja + 4);
        int4 k2 = *(const int4*)(mrow + jb);
        int4 k3 = *(const int4*)(mrow + jb + 4);

        float ev[16];
        ev[0]=si+d0.x;  ev[1]=si+d0.y;  ev[2]=si+d0.z;  ev[3]=si+d0.w;
        ev[4]=si+d1.x;  ev[5]=si+d1.y;  ev[6]=si+d1.z;  ev[7]=si+d1.w;
        ev[8]=si+d2.x;  ev[9]=si+d2.y;  ev[10]=si+d2.z; ev[11]=si+d2.w;
        ev[12]=si+d3.x; ev[13]=si+d3.y; ev[14]=si+d3.z; ev[15]=si+d3.w;
        int mk[16] = {k0.x,k0.y,k0.z,k0.w, k1.x,k1.y,k1.z,k1.w,
                      k2.x,k2.y,k2.z,k2.w, k3.x,k3.y,k3.z,k3.w};
#pragma unroll
        for (int t = 0; t < 16; ++t) {
            float v = ev[t];
            v = v > 0.f ? v : ALPHA * v;               // leaky relu
            ev[t] = (mk[t] != 0) ? v : -1.0e30f;       // mask
        }

        float tmax = ev[0];
#pragma unroll
        for (int t = 1; t < 16; ++t) tmax = fmaxf(tmax, ev[t]);
        tmax = fmaxf(tmax, __shfl_xor(tmax, 16, 32));
        float mnew  = fmaxf(m, tmax);
        float scale = __expf(m - mnew);

        float psum = 0.f;
        v16bf P;
#pragma unroll
        for (int t = 0; t < 16; ++t) {
            float p = __expf(ev[t] - mnew);
            psum += p;
            P[t] = (__bf16)p;
        }
        psum += __shfl_xor(psum, 16, 32);
        sum = sum * scale + psum;
        m = mnew;

        float sc[8];
#pragma unroll
        for (int r = 0; r < 8; ++r) sc[r] = __shfl(scale, hi * 8 + r, 32);
#pragma unroll
        for (int r = 0; r < 8; ++r) {
            acc0[r] *= sc[r]; acc1[r] *= sc[r];
            acc2[r] *= sc[r]; acc3[r] *= sc[r];
        }

        // B tiles from LDS: rows j0..j0+31, 4 chunks of 16 f-columns
        const char* bp = sS + buf * (64 * COLB) + l * COLB + hi * 32;
        v16bf B0 = *(const v16bf*)(bp + 0 * 16 * COLB);
        v16bf B1 = *(const v16bf*)(bp + 1 * 16 * COLB);
        v16bf B2 = *(const v16bf*)(bp + 2 * 16 * COLB);
        v16bf B3 = *(const v16bf*)(bp + 3 * 16 * COLB);

        acc0 = __builtin_amdgcn_wmma_f32_16x16x32_bf16(false, P, false, B0, (short)0, acc0, false, false);
        acc1 = __builtin_amdgcn_wmma_f32_16x16x32_bf16(false, P, false, B1, (short)0, acc1, false, false);
        acc2 = __builtin_amdgcn_wmma_f32_16x16x32_bf16(false, P, false, B2, (short)0, acc2, false, false);
        acc3 = __builtin_amdgcn_wmma_f32_16x16x32_bf16(false, P, false, B3, (short)0, acc3, false, false);

        wait_async();
        __syncthreads();
        buf ^= 1;
    }

    float li[8];
#pragma unroll
    for (int r = 0; r < 8; ++r) li[r] = 1.f / __shfl(sum, hi * 8 + r, 32);

    v8f accs[4] = {acc0, acc1, acc2, acc3};
#pragma unroll
    for (int c = 0; c < 4; ++c) {
#pragma unroll
        for (int r = 0; r < 8; ++r) {
            int i = itile * 16 + hi * 8 + r;
            out[((size_t)(b * SEQ + i)) * NCOL + h * FDIM + c * 16 + l] = accs[c][r] * li[r];
        }
    }
}

// ---------------------------------------------------------------------------
extern "C" void kernel_launch(void* const* d_in, const int* in_sizes, int n_in,
                              void* d_out, int out_size, void* d_ws, size_t ws_size,
                              hipStream_t stream) {
    const float* X    = (const float*)d_in[0];   // (8,1024,768) f32
    const int*   mask = (const int*)  d_in[1];   // (8,1024) i32
    const float* W    = (const float*)d_in[2];   // (768,512) f32
    const float* avec = (const float*)d_in[3];   // (128,) f32
    float* out = (float*)d_out;                  // (8,1024,512) f32

    char* ws = (char*)d_ws;
    __bf16* WT   = (__bf16*)(ws);                          // 786432 B
    __bf16* WhT  = (__bf16*)(ws + 786432);                 // 8388608 B
    float*  srcb = (float*) (ws + 786432 + 8388608);       // 262144 B
    float*  dstb = (float*) (ws + 786432 + 8388608 + 262144);

    k_wt<<<(INDIM * NCOL) / 256, 256, 0, stream>>>(W, WT);
    k_gemm<<<512, 256, 0, stream>>>(X, WT, WhT);
    k_srcdst<<<(BATCH * NHEAD * SEQ) / 256, 256, 0, stream>>>(WhT, avec, srcb, dstb);
    k_attn<<<512, 256, 0, stream>>>(WhT, srcb, dstb, mask, out);
}
